// UPFDGraphSageNet_69758858821965
// MI455X (gfx1250) — matrix operand order repaired and checked by reading, not compiled, not run on hardware
//
#include <hip/hip_runtime.h>
#include <hip/hip_bf16.h>

// ---------------------------------------------------------------------------
// Problem constants (match reference)
// ---------------------------------------------------------------------------
constexpr int NN   = 50000;    // nodes
constexpr int NE   = 600000;   // edges
constexpr int CIN  = 300;      // input channels
constexpr int KP1  = 320;      // layer-1 K padded to multiple of 32
constexpr int H    = 128;      // hidden
constexpr int NG   = 128;      // graphs
constexpr int MPAD = 50048;    // nodes padded to multiple of 128 (391*128)
constexpr float EPS = 1e-5f;

constexpr int NT      = H / 16;       // 8 column tiles
constexpr int TILE_SH = NT * 512;     // shorts per swizzled weight K-tile (8 KB)

// ---------------------------------------------------------------------------
// CDNA5 async global->LDS path (guarded; fallback = load + ds_store)
// ---------------------------------------------------------------------------
#ifndef __has_builtin
#define __has_builtin(x) 0
#endif
#if __has_builtin(__builtin_amdgcn_global_load_async_to_lds_b128)
#define USE_ASYNC_LDS 1
#else
#define USE_ASYNC_LDS 0
#endif

typedef int v4i __attribute__((vector_size(16)));   // matches builtin's b128 type

__device__ __forceinline__ void wait_async_lds() {
#if USE_ASYNC_LDS
#if __has_builtin(__builtin_amdgcn_s_wait_asynccnt)
    __builtin_amdgcn_s_wait_asynccnt(0);
#else
    asm volatile("s_wait_asynccnt 0" ::: "memory");
#endif
#endif
}

// ---------------------------------------------------------------------------
// WMMA fragment types (CDNA5 wave32, 16x16x32 bf16)
// ---------------------------------------------------------------------------
typedef __attribute__((ext_vector_type(16))) __bf16     v16bf;
typedef __attribute__((ext_vector_type(8)))  float      v8f;
typedef __attribute__((ext_vector_type(8)))  unsigned short ushort8;

union FragAB {
    ushort8 h[2];   // two 16-byte halves
    v16bf   v;      // 16 bf16 elements = 8 VGPRs
};

__device__ __forceinline__ unsigned short f2bf(float f) {
    unsigned int u = __builtin_bit_cast(unsigned int, f);
    u += 0x7fffu + ((u >> 16) & 1u);       // round-to-nearest-even
    return (unsigned short)(u >> 16);
}

// ---------------------------------------------------------------------------
// Weight swizzle: W[K][128] fp32 -> bf16 fragment layout [kt][nt][lane][16]
// lane half (lane>>4) selects K = kt*32 + 8*half + e (e<8), kt*32 + 16 + 8*half + (e-8) (e>=8) ; N = nt*16 + (lane&15)
// ---------------------------------------------------------------------------
__global__ void swizzle_w_kernel(const float* __restrict__ W,
                                 unsigned short* __restrict__ Wswz,
                                 int K, int Kpad, int Ncols) {
    const int nt_cnt = Ncols >> 4;
    const int total = (Kpad >> 5) * nt_cnt * 512;
    int idx = blockIdx.x * blockDim.x + threadIdx.x;
    if (idx >= total) return;
    int e    = idx & 15;
    int lane = (idx >> 4) & 31;
    int ntkt = idx >> 9;
    int nt   = ntkt % nt_cnt;
    int kt   = ntkt / nt_cnt;
    int k = kt * 32 + (lane >> 4) * 8 + (e & 7) + ((e >> 3) << 4);
    int n = nt * 16 + (lane & 15);
    float v = (k < K) ? W[(size_t)k * Ncols + n] : 0.f;
    Wswz[idx] = f2bf(v);
}

// x fp32 [NN][CIN] -> bf16 [MPAD][KP1] (pad region pre-zeroed by memset)
__global__ void convert_x_kernel(const float* __restrict__ x,
                                 unsigned short* __restrict__ Abf) {
    int idx = blockIdx.x * blockDim.x + threadIdx.x;
    if (idx >= NN * CIN) return;
    int r = idx / CIN, c = idx % CIN;
    Abf[(size_t)r * KP1 + c] = f2bf(x[idx]);
}

__global__ void degree_kernel(const int* __restrict__ dst,
                              float* __restrict__ deg, int E) {
    int e = blockIdx.x * blockDim.x + threadIdx.x;
    if (e < E) atomicAdd(&deg[dst[e]], 1.0f);
}

// one wave per edge; float4 gather of the source row, 4 atomics per chunk
__global__ __launch_bounds__(32)
void scatter_add_kernel(const int* __restrict__ src, const int* __restrict__ dst,
                        const float* __restrict__ act, float* __restrict__ agg,
                        int E, int C4) {
    int e = blockIdx.x;
    if (e >= E) return;
    const float4* a = (const float4*)(act + (size_t)src[e] * (C4 * 4));
    float*        o = agg + (size_t)dst[e] * (C4 * 4);
    for (int c = threadIdx.x; c < C4; c += 32) {
        float4 v = a[c];
        atomicAdd(o + c * 4 + 0, v.x);
        atomicAdd(o + c * 4 + 1, v.y);
        atomicAdd(o + c * 4 + 2, v.z);
        atomicAdd(o + c * 4 + 3, v.w);
    }
}

// mean = agg / max(deg,1) -> bf16 padded operand
__global__ void finalize_agg_kernel(const float* __restrict__ agg,
                                    const float* __restrict__ deg,
                                    unsigned short* __restrict__ Abf,
                                    int C, int Kpad) {
    int idx = blockIdx.x * blockDim.x + threadIdx.x;
    if (idx >= NN * C) return;
    int r = idx / C, c = idx % C;
    float d = fmaxf(deg[r], 1.0f);
    Abf[(size_t)r * Kpad + c] = f2bf(agg[idx] / d);
}

// ---------------------------------------------------------------------------
// Stage one swizzled weight K-tile (Wl || Wr, 16 KB) into LDS.
// 128 threads x 8 chunks of 16 B each; async path uses GLOBAL_LOAD_ASYNC_TO_LDS.
// ---------------------------------------------------------------------------
__device__ __forceinline__ void stage_tile(const unsigned short* __restrict__ gl,
                                           const unsigned short* __restrict__ gr,
                                           unsigned short* sbuf, int tid) {
#pragma unroll
    for (int i = 0; i < 8; ++i) {
        int chunk = i * 128 + tid;            // 1024 chunks of 8 shorts
        int el    = chunk * 8;
        const unsigned short* g = (el < TILE_SH) ? (gl + el) : (gr + (el - TILE_SH));
#if USE_ASYNC_LDS
        __builtin_amdgcn_global_load_async_to_lds_b128(
            (__attribute__((address_space(1))) v4i*)(g),
            (__attribute__((address_space(3))) v4i*)(sbuf + el),
            0, 0);
#else
        *(ushort8*)(sbuf + el) = *(const ushort8*)g;
#endif
    }
}

__device__ __forceinline__ void load_b_frag(const unsigned short* p, FragAB& f) {
    f.h[0] = *(const ushort8*)(p);
    f.h[1] = *(const ushort8*)(p + 8);
}

// ---------------------------------------------------------------------------
// Fused dual GEMM: out[M][128] = Am @ Wl + Ax @ Wr + bias
// 4 waves/block; each wave owns a 32x128 strip (two stacked 16-row M tiles so
// every B fragment feeds two WMMAs). Weight K-tiles are double-buffered in LDS
// via async global->LDS; inner loop software-pipelines the ds_load_b128 B
// fragments one nt ahead so LDS latency hides under in-flight WMMAs.
// A fragment layout (16-bit A 16x32): lane M = lane&15, half = lane>>4,
//   elements 0-7  = K kb + 8*half + 0..7 ; elements 8-15 = K kb + 16 + 8*half
// ---------------------------------------------------------------------------
__global__ __launch_bounds__(128)
void gemm_dual_wmma_kernel(const unsigned short* __restrict__ Am,
                           const unsigned short* __restrict__ Ax,
                           const unsigned short* __restrict__ Wl,
                           const unsigned short* __restrict__ Wr,
                           const float* __restrict__ bias,
                           float* __restrict__ out, int Kpad) {
    __shared__ alignas(16) unsigned short smem[2][2 * TILE_SH];   // 32 KB

    const int tid  = threadIdx.x;
    const int wave = tid >> 5;
    const int lane = tid & 31;
    const int half = lane >> 4;
    const int l15  = lane & 15;
    const int mtileA = blockIdx.x * 8 + wave * 2;   // this wave: tiles mtileA, mtileA+1
    const int rowA   = mtileA * 16 + l15;

    v8f accA[NT], accB[NT];
#pragma unroll
    for (int t = 0; t < NT; ++t) {
        accA[t] = (v8f){0,0,0,0,0,0,0,0};
        accB[t] = (v8f){0,0,0,0,0,0,0,0};
    }

    const unsigned short* amRow = Am + (size_t)rowA * Kpad;
    const unsigned short* axRow = Ax + (size_t)rowA * Kpad;
    const size_t rowStride16 = (size_t)16 * Kpad;
    const int KT = Kpad >> 5;

    stage_tile(Wl, Wr, smem[0], tid);

    for (int kt = 0; kt < KT; ++kt) {
        unsigned short* sbuf = smem[kt & 1];
        wait_async_lds();
        __syncthreads();                       // staged tile visible to all waves
        if (kt + 1 < KT)
            stage_tile(Wl + (size_t)(kt + 1) * TILE_SH,
                       Wr + (size_t)(kt + 1) * TILE_SH,
                       smem[(kt + 1) & 1], tid);

        const int kb = kt * 32;
        FragAB amA, amB, axA, axB;
        amA.h[0] = *(const ushort8*)(amRow + kb + half * 8);
        amA.h[1] = *(const ushort8*)(amRow + kb + 16 + half * 8);
        axA.h[0] = *(const ushort8*)(axRow + kb + half * 8);
        axA.h[1] = *(const ushort8*)(axRow + kb + 16 + half * 8);
        amB.h[0] = *(const ushort8*)(amRow + rowStride16 + kb + half * 8);
        amB.h[1] = *(const ushort8*)(amRow + rowStride16 + kb + 16 + half * 8);
        axB.h[0] = *(const ushort8*)(axRow + rowStride16 + kb + half * 8);
        axB.h[1] = *(const ushort8*)(axRow + rowStride16 + kb + 16 + half * 8);

        const unsigned short* sl = sbuf + lane * 16;            // Wl tile
        const unsigned short* sr = sbuf + TILE_SH + lane * 16;  // Wr tile

        // software-pipelined B fragments: preload nt+1 before WMMAs of nt
        FragAB bl[2], br[2];
        load_b_frag(sl, bl[0]);
        load_b_frag(sr, br[0]);
#pragma unroll
        for (int nt = 0; nt < NT; ++nt) {
            const int cur = nt & 1, nxt = cur ^ 1;
            if (nt + 1 < NT) {
                load_b_frag(sl + (nt + 1) * 512, bl[nxt]);
                load_b_frag(sr + (nt + 1) * 512, br[nxt]);
            }
            accA[nt] = __builtin_amdgcn_wmma_f32_16x16x32_bf16(
                false, amA.v, false, bl[cur].v, (short)0, accA[nt], false, false);
            accB[nt] = __builtin_amdgcn_wmma_f32_16x16x32_bf16(
                false, amB.v, false, bl[cur].v, (short)0, accB[nt], false, false);
            accA[nt] = __builtin_amdgcn_wmma_f32_16x16x32_bf16(
                false, axA.v, false, br[cur].v, (short)0, accA[nt], false, false);
            accB[nt] = __builtin_amdgcn_wmma_f32_16x16x32_bf16(
                false, axB.v, false, br[cur].v, (short)0, accB[nt], false, false);
        }
        __syncthreads();                       // done reading sbuf
    }

    // C/D layout: VGPR v -> row mtile*16 + half*8 + v, col = nt*16 + (lane&15)
    const int baseRow = mtileA * 16 + half * 8;
#pragma unroll
    for (int nt = 0; nt < NT; ++nt) {
        const int col = nt * 16 + l15;
        const float b = bias[col];
#pragma unroll
        for (int v = 0; v < 8; ++v) {
            out[(size_t)(baseRow + v) * H + col]      = accA[nt][v] + b;
            out[(size_t)(baseRow + 16 + v) * H + col] = accB[nt][v] + b;
        }
    }
}

// ---------------------------------------------------------------------------
// BN statistics per column (deterministic tree reduce), folded into scale/shift
// ---------------------------------------------------------------------------
__global__ __launch_bounds__(256)
void colstats_kernel(const float* __restrict__ h,
                     const float* __restrict__ gamma,
                     const float* __restrict__ beta,
                     float* __restrict__ scale, float* __restrict__ shift) {
    __shared__ float sh1[256], sh2[256];
    const int col = blockIdx.x;
    float s = 0.f, s2 = 0.f;
    for (int r = threadIdx.x; r < NN; r += 256) {
        float v = h[(size_t)r * H + col];
        s += v; s2 += v * v;
    }
    sh1[threadIdx.x] = s; sh2[threadIdx.x] = s2;
    __syncthreads();
    for (int o = 128; o > 0; o >>= 1) {
        if (threadIdx.x < o) {
            sh1[threadIdx.x] += sh1[threadIdx.x + o];
            sh2[threadIdx.x] += sh2[threadIdx.x + o];
        }
        __syncthreads();
    }
    if (threadIdx.x == 0) {
        float mu  = sh1[0] / (float)NN;
        float var = sh2[0] / (float)NN - mu * mu;
        float sc  = gamma[col] * rsqrtf(var + EPS);
        scale[col] = sc;
        shift[col] = beta[col] - mu * sc;
    }
}

// h_act = relu(h*scale+shift); also emit bf16 operand for the next GEMM
__global__ void bnrelu_kernel(const float* __restrict__ hp,
                              const float* __restrict__ scale,
                              const float* __restrict__ shift,
                              float* __restrict__ hact,
                              unsigned short* __restrict__ Abf) {
    int idx = blockIdx.x * blockDim.x + threadIdx.x;
    if (idx >= NN * H) return;
    int c = idx & (H - 1);
    float a = fmaxf(hp[idx] * scale[c] + shift[c], 0.f);
    hact[idx] = a;
    Abf[idx]  = f2bf(a);
}

__global__ void pool_scatter_kernel(const float* __restrict__ h,
                                    const int* __restrict__ batch,
                                    float* __restrict__ pooled,
                                    float* __restrict__ cnt) {
    int idx = blockIdx.x * blockDim.x + threadIdx.x;
    if (idx >= NN * H) return;
    int r = idx >> 7, c = idx & (H - 1);
    int g = batch[r];
    atomicAdd(&pooled[(size_t)g * H + c], h[idx]);
    if (c == 0) atomicAdd(&cnt[g], 1.0f);
}

__global__ void pool_final_kernel(const float* __restrict__ pooled,
                                  const float* __restrict__ cnt,
                                  float* __restrict__ hg) {
    int idx = blockIdx.x * blockDim.x + threadIdx.x;
    if (idx >= NG * H) return;
    hg[idx] = pooled[idx] / fmaxf(cnt[idx >> 7], 1.0f);
}

__global__ void classify_kernel(const float* __restrict__ hg,
                                const float* __restrict__ Wc,
                                const float* __restrict__ bc,
                                float* __restrict__ out) {
    int t = threadIdx.x;
    if (t >= NG * 2) return;
    int g = t >> 1, c = t & 1;
    float s = bc[c];
    for (int k = 0; k < H; ++k)
        s += hg[(size_t)g * H + k] * Wc[k * 2 + c];
    out[g * 2 + c] = s;
}

// ---------------------------------------------------------------------------
// Host side
// ---------------------------------------------------------------------------
extern "C" void kernel_launch(void* const* d_in, const int* in_sizes, int n_in,
                              void* d_out, int out_size, void* d_ws, size_t ws_size,
                              hipStream_t stream) {
    (void)in_sizes; (void)n_in; (void)out_size; (void)ws_size;

    const float* x     = (const float*)d_in[0];
    const int*   ei    = (const int*)d_in[1];
    const int*   batch = (const int*)d_in[2];
    const float* Wl_[3] = {(const float*)d_in[3],  (const float*)d_in[8],  (const float*)d_in[13]};
    const float* Wr_[3] = {(const float*)d_in[4],  (const float*)d_in[9],  (const float*)d_in[14]};
    const float* b_[3]  = {(const float*)d_in[5],  (const float*)d_in[10], (const float*)d_in[15]};
    const float* g_[3]  = {(const float*)d_in[6],  (const float*)d_in[11], (const float*)d_in[16]};
    const float* be_[3] = {(const float*)d_in[7],  (const float*)d_in[12], (const float*)d_in[17]};
    const float* Wc = (const float*)d_in[18];
    const float* bc = (const float*)d_in[19];
    const int* src = ei;
    const int* dst = ei + NE;

    // ---- workspace carve-out (256B aligned) ----
    char* ws = (char*)d_ws;
    size_t off = 0;
    auto carve = [&](size_t bytes) -> char* {
        char* p = ws + off;
        off = (off + bytes + 255) & ~(size_t)255;
        return p;
    };
    float*          deg   = (float*)carve((size_t)NN * 4);
    float*          agg   = (float*)carve((size_t)NN * CIN * 4);       // reused at C=128
    unsigned short* Ax1   = (unsigned short*)carve((size_t)MPAD * KP1 * 2);
    unsigned short* Am1   = (unsigned short*)carve((size_t)MPAD * KP1 * 2);
    unsigned short* Ax2   = (unsigned short*)carve((size_t)MPAD * H * 2);
    unsigned short* Am2   = (unsigned short*)carve((size_t)MPAD * H * 2);
    const size_t swz1 = (size_t)(KP1 / 32) * TILE_SH;                  // elems
    const size_t swz2 = (size_t)(H   / 32) * TILE_SH;
    unsigned short* Wls[3], *Wrs[3];
    Wls[0] = (unsigned short*)carve(swz1 * 2);
    Wrs[0] = (unsigned short*)carve(swz1 * 2);
    Wls[1] = (unsigned short*)carve(swz2 * 2);
    Wrs[1] = (unsigned short*)carve(swz2 * 2);
    Wls[2] = (unsigned short*)carve(swz2 * 2);
    Wrs[2] = (unsigned short*)carve(swz2 * 2);
    float* hpre   = (float*)carve((size_t)MPAD * H * 4);
    float* hact   = (float*)carve((size_t)NN * H * 4);
    float* scale  = (float*)carve(H * 4);
    float* shift  = (float*)carve(H * 4);
    float* pooled = (float*)carve((size_t)NG * H * 4);
    float* cnt    = (float*)carve((size_t)NG * 4);

    // ---- re-zero everything we accumulate into / pad regions ----
    (void)hipMemsetAsync(deg,    0, (size_t)NN * 4, stream);
    (void)hipMemsetAsync(agg,    0, (size_t)NN * CIN * 4, stream);
    (void)hipMemsetAsync(Ax1,    0, (size_t)MPAD * KP1 * 2, stream);
    (void)hipMemsetAsync(Am1,    0, (size_t)MPAD * KP1 * 2, stream);
    (void)hipMemsetAsync(Ax2,    0, (size_t)MPAD * H * 2, stream);
    (void)hipMemsetAsync(Am2,    0, (size_t)MPAD * H * 2, stream);
    (void)hipMemsetAsync(pooled, 0, (size_t)NG * H * 4, stream);
    (void)hipMemsetAsync(cnt,    0, (size_t)NG * 4, stream);

    // ---- weight prep (bf16 fragment layout) ----
    {
        int tot1 = (int)swz1;
        int tot2 = (int)swz2;
        swizzle_w_kernel<<<(tot1 + 255) / 256, 256, 0, stream>>>(Wl_[0], Wls[0], CIN, KP1, H);
        swizzle_w_kernel<<<(tot1 + 255) / 256, 256, 0, stream>>>(Wr_[0], Wrs[0], CIN, KP1, H);
        for (int l = 1; l < 3; ++l) {
            swizzle_w_kernel<<<(tot2 + 255) / 256, 256, 0, stream>>>(Wl_[l], Wls[l], H, H, H);
            swizzle_w_kernel<<<(tot2 + 255) / 256, 256, 0, stream>>>(Wr_[l], Wrs[l], H, H, H);
        }
    }
    convert_x_kernel<<<(NN * CIN + 255) / 256, 256, 0, stream>>>(x, Ax1);
    degree_kernel<<<(NE + 255) / 256, 256, 0, stream>>>(dst, deg, NE);

    const int gemmBlocks = MPAD / 128;   // 391

    // ---- layer 1 (K = 300 -> 320) ----
    scatter_add_kernel<<<NE, 32, 0, stream>>>(src, dst, x, agg, NE, CIN / 4);
    finalize_agg_kernel<<<(NN * CIN + 255) / 256, 256, 0, stream>>>(agg, deg, Am1, CIN, KP1);
    gemm_dual_wmma_kernel<<<gemmBlocks, 128, 0, stream>>>(Am1, Ax1, Wls[0], Wrs[0], b_[0], hpre, KP1);
    colstats_kernel<<<H, 256, 0, stream>>>(hpre, g_[0], be_[0], scale, shift);
    bnrelu_kernel<<<(NN * H + 255) / 256, 256, 0, stream>>>(hpre, scale, shift, hact, Ax2);

    // ---- layers 2,3 (K = 128) ----
    for (int l = 1; l < 3; ++l) {
        (void)hipMemsetAsync(agg, 0, (size_t)NN * H * 4, stream);
        scatter_add_kernel<<<NE, 32, 0, stream>>>(src, dst, hact, agg, NE, H / 4);
        finalize_agg_kernel<<<(NN * H + 255) / 256, 256, 0, stream>>>(agg, deg, Am2, H, H);
        gemm_dual_wmma_kernel<<<gemmBlocks, 128, 0, stream>>>(Am2, Ax2, Wls[l], Wrs[l], b_[l], hpre, H);
        colstats_kernel<<<H, 256, 0, stream>>>(hpre, g_[l], be_[l], scale, shift);
        bnrelu_kernel<<<(NN * H + 255) / 256, 256, 0, stream>>>(hpre, scale, shift, hact, Ax2);
    }

    // ---- pooling + classifier; d_out = [out(128x2) | hg(128x128)] ----
    float* out_p = (float*)d_out;
    float* hg_p  = out_p + NG * 2;
    pool_scatter_kernel<<<(NN * H + 255) / 256, 256, 0, stream>>>(hact, batch, pooled, cnt);
    pool_final_kernel<<<(NG * H + 255) / 256, 256, 0, stream>>>(pooled, cnt, hg_p);
    classify_kernel<<<1, 256, 0, stream>>>(hg_p, Wc, bc, out_p);
}
